// ContrastiveCRFLoss_22969485099580
// MI455X (gfx1250) — matrix operand order repaired and checked by reading, not compiled, not run on hardware
//
#include <hip/hip_runtime.h>
#include <hip/hip_bf16.h>

typedef __attribute__((ext_vector_type(16))) _Float16 v16h;
typedef __attribute__((ext_vector_type(8)))  _Float16 v8h;
typedef __attribute__((ext_vector_type(8)))  float    v8f;

#define NSAMP 1000
#define NPAD  1008        // 63 * 16
#define NT    63
#define NTFULL 62         // tiles 0..61 are fully in-range (16*62 = 992 <= 1000)
#define BATCH 32
#define KCL   27
#define KPAD  32
#define NG    3
#define GSTRIDE (BATCH * NG)   // 96 floats per sample row in selg
#define HW    65536       // 256*256
#define W_IMG 256

// ---------------------------------------------------------------------------
// Phase 1: gather guidance / clusters at the sampled coords into packed,
// WMMA-friendly layouts in workspace.
//   rf[1008], cf[1008]                  : sample coords as float (zero padded)
//   selg[s][n*3+g]   (f32, 1008 x 96)   : guidance at samples
//   selc[n][s][k]    (f16, 32x1008x32)  : clusters at samples, K padded 27->32
// ---------------------------------------------------------------------------
__global__ __launch_bounds__(256) void crf_gather(
    const float* __restrict__ guidance,
    const float* __restrict__ clusters,
    const int*   __restrict__ coords,
    float* __restrict__ rf, float* __restrict__ cfv,
    float* __restrict__ selg, _Float16* __restrict__ selc)
{
  int idx = blockIdx.x * blockDim.x + threadIdx.x;
  if (idx >= BATCH * NPAD) return;
  int n = idx / NPAD;
  int s = idx - n * NPAD;
  bool valid = (s < NSAMP);
  int r = valid ? coords[s] : 0;
  int c = valid ? coords[NSAMP + s] : 0;
  if (n == 0) {
    rf[s]  = valid ? (float)r : 0.0f;
    cfv[s] = valid ? (float)c : 0.0f;
  }
  int pix = r * W_IMG + c;
  float* gout = selg + (size_t)s * GSTRIDE + n * NG;
#pragma unroll
  for (int g = 0; g < NG; ++g)
    gout[g] = valid ? guidance[(size_t)(n * NG + g) * HW + pix] : 0.0f;
  _Float16* co = selc + ((size_t)n * NPAD + s) * KPAD;
#pragma unroll
  for (int k = 0; k < KPAD; ++k)
    co[k] = (_Float16)((valid && k < KCL)
                           ? clusters[(size_t)(n * KCL + k) * HW + pix]
                           : 0.0f);
}

// Load one wave's A/B WMMA fragments for batch-slab `cbase`
// (contiguous K order: elems 0..15 of lane half h hold K = 16h..16h+15).
__device__ __forceinline__ void load_frags(
    const _Float16* __restrict__ cbase,
    int sa, int klo, int sb, int kb, v16h& af, v16h& bf)
{
  const _Float16* ap = cbase + (size_t)sa * KPAD + klo;
  v8h alo = *(const v8h*)(ap);
  v8h ahi = *(const v8h*)(ap + 8);
  af = __builtin_shufflevector(alo, ahi,
        0,1,2,3,4,5,6,7,8,9,10,11,12,13,14,15);
  bf = *(const v16h*)(cbase + (size_t)sb * KPAD + kb);
}

// Batch loop, software-pipelined one stage: batch n+1's fragments + WMMA are
// issued before batch n's accumulator is consumed, so the WMMA->VALU RAW
// hazard slots are filled with guidance/exp math instead of v_nops.
// Unrolled by 2 so the two accumulators ping-pong without v_mov rotates.
template <bool FULL>
__device__ __forceinline__ void crf_batches(
    const float* __restrict__ selg, const _Float16* __restrict__ selc,
    float* __restrict__ out,
    int sa, int klo, int sb, int kb, int ag0, int bg, bool bok,
    const float* cdm, const float* E2)
{
  const float LOG2E = 1.44269504088896340736f;
  const float GK    = -LOG2E * (1.0f / 0.3f);        // guidance term coeff

  v16h af, bf;
  load_frags(selc, sa, klo, sb, kb, af, bf);
  v8f accP = __builtin_amdgcn_wmma_f32_16x16x32_f16(
      false, af, false, bf, (short)0, (v8f){}, false, false);

#pragma unroll 2
  for (int n = 0; n < BATCH; ++n) {
    // Prefetch + issue next batch's WMMA (independent of accP).
    if (n < BATCH - 1) {
      const _Float16* cnext = selc + (size_t)(n + 1) * NPAD * KPAD;
      load_frags(cnext, sa, klo, sb, kb, af, bf);
    }

    // Guidance values for batch n: one base pointer + const offsets (v*96).
    const float* gcol = selg + (size_t)bg  * GSTRIDE + n * NG;
    const float* grow = selg + (size_t)ag0 * GSTRIDE + n * NG;
    float gb0 = gcol[0], gb1 = gcol[1], gb2 = gcol[2];
    float g0[8], g1[8], g2[8];
#pragma unroll
    for (int v = 0; v < 8; ++v) {
      g0[v] = grow[v * GSTRIDE + 0];
      g1[v] = grow[v * GSTRIDE + 1];
      g2[v] = grow[v * GSTRIDE + 2];
    }

    v8f accC = accP;
    if (n < BATCH - 1)
      accP = __builtin_amdgcn_wmma_f32_16x16x32_f16(
          false, af, false, bf, (short)0, (v8f){}, false, false);

    float val[8];
#pragma unroll
    for (int v = 0; v < 8; ++v) {
      float d0 = g0[v] - gb0;
      float d1 = g1[v] - gb1;
      float d2 = g2[v] - gb2;
      float gd  = fmaf(d2, d2, fmaf(d1, d1, d0 * d0));
      float sim = fmaf(10.0f,
                       __builtin_amdgcn_exp2f(fmaf(gd, GK, cdm[v])),
                       E2[v]);
      val[v] = -accC[v] * sim;
    }

    float* obase = out + (size_t)n * (NSAMP * NSAMP)
                       + (size_t)ag0 * NSAMP + bg;
    if (FULL) {
#pragma unroll
      for (int v = 0; v < 8; ++v)
        obase[v * NSAMP] = val[v];
    } else {
#pragma unroll
      for (int v = 0; v < 8; ++v)
        if (bok && (ag0 + v) < NSAMP)
          obase[v * NSAMP] = val[v];
    }
  }
}

// ---------------------------------------------------------------------------
// Phase 2: one wave per 16x16 (a,b) output tile; coordinate kernel terms
// precomputed once per tile, then all 32 batches processed reusing them.
// ---------------------------------------------------------------------------
__global__ __launch_bounds__(256) void crf_main(
    const float* __restrict__ rf, const float* __restrict__ cfv,
    const float* __restrict__ selg, const _Float16* __restrict__ selc,
    float* __restrict__ out)
{
  // Force the tile id into an SGPR: it is uniform within the wave by
  // construction, and this makes ta/tb/full scalar so the interior/edge
  // split is a scalar branch (no EXEC masking in the hot loop).
  const int tile = __builtin_amdgcn_readfirstlane(
      (int)(blockIdx.x * 8 + (threadIdx.x >> 5)));
  if (tile >= NT * NT) return;                       // scalar exit
  const int ta   = tile / NT;
  const int tb   = tile - ta * NT;
  const int lane = threadIdx.x & 31;
  const int half = lane >> 4;
  const int lm   = lane & 15;

  const int sa  = ta * 16 + lm;    // A-matrix row sample for this lane
  const int klo = half * 16;       // A elems 0..15 hold K=klo..klo+15
  const int sb  = tb * 16 + lm;    // B-matrix column sample (= output column)
  const int kb  = half * 16;       // B elems 0..15 hold K=kb..kb+15

  const float LOG2E = 1.44269504088896340736f;

  const float rb = rf[sb];
  const float cb = cfv[sb];
  const int   bg = sb;                               // global column index
  const bool  bok = (bg < NSAMP);
  const int   ag0 = ta * 16 + 8 * half;              // first of 8 C/D rows

  // Batch-independent coordinate kernel terms, computed once per tile.
  float cdm[8], E2[8];
#pragma unroll
  for (int v = 0; v < 8; ++v) {
    float dr = rf[ag0 + v]  - rb;
    float dc = cfv[ag0 + v] - cb;
    float cd = fmaf(dr, dr, dc * dc);
    cdm[v] = -LOG2E * cd;                            // -cd/(2*ALPHA) in log2
    E2[v]  = 3.0f * __builtin_amdgcn_exp2f(-10.0f * LOG2E * cd); // W2 term
  }

  if ((ta < NTFULL) && (tb < NTFULL)) {              // scalar branch
    crf_batches<true >(selg, selc, out, sa, klo, sb, kb, ag0, bg, bok, cdm, E2);
  } else {
    crf_batches<false>(selg, selc, out, sa, klo, sb, kb, ag0, bg, bok, cdm, E2);
  }
}

extern "C" void kernel_launch(void* const* d_in, const int* in_sizes, int n_in,
                              void* d_out, int out_size, void* d_ws, size_t ws_size,
                              hipStream_t stream) {
  const float* guidance = (const float*)d_in[0];   // (32,3,256,256) f32
  const float* clusters = (const float*)d_in[1];   // (32,27,256,256) f32
  const int*   coords   = (const int*)d_in[2];     // (2,1000) i32
  float* out = (float*)d_out;                      // (32,1000,1000) f32

  // Workspace layout (<= ~2.4 MB): rf | cf | selg | selc (32B-aligned)
  float*    rf   = (float*)d_ws;
  float*    cfv  = rf + NPAD;
  float*    selg = cfv + NPAD;
  _Float16* selc = (_Float16*)(selg + (size_t)NPAD * GSTRIDE);

  {
    int total = BATCH * NPAD;                      // 32256
    dim3 grid((total + 255) / 256), block(256);
    crf_gather<<<grid, block, 0, stream>>>(guidance, clusters, coords,
                                           rf, cfv, selg, selc);
  }
  {
    int tiles = NT * NT;                           // 3969 waves
    dim3 grid((tiles + 7) / 8), block(256);        // 8 waves per block
    crf_main<<<grid, block, 0, stream>>>(rf, cfv, selg, selc, out);
  }
}
